// QuantLinear_23270132809729
// MI455X (gfx1250) — matrix operand order repaired and checked by reading, not compiled, not run on hardware
//
#include <hip/hip_runtime.h>

typedef __attribute__((ext_vector_type(16))) _Float16 v16h;
typedef __attribute__((ext_vector_type(8)))  _Float16 v8h;
typedef __attribute__((ext_vector_type(4)))  _Float16 v4h;
typedef __attribute__((ext_vector_type(2)))  _Float16 v2h;
typedef __attribute__((ext_vector_type(8)))  float    v8f;

#define BLOCK_THREADS   256
#define WAVES_PER_BLOCK 8
#define TILES_PER_WAVE  2      // 2 x 16 output columns per wave
#define MAX_SPLIT_K     8

// Magic-bias dequant: two nibbles (m, m+4) of w -> two f16 values in one
// v_pk_add_f16.  0x6400 = f16(1024); 1024+n is integer-exact in f16.
__device__ __forceinline__ v2h deq2(unsigned w, int m) {
  const unsigned t = ((w >> (4 * m)) & 0x000F000Fu) | 0x64006400u;
  const v2h bias = {(_Float16)-1024.0f, (_Float16)-1024.0f};
  return __builtin_bit_cast(v2h, t) + bias;
}

// B fragment halfs come out in permuted K order sigma = (0,4,1,5,2,6,3,7)
// within each aligned 8-group; the x slice is staged into LDS with the same
// permutation, so the WMMA dot product is unchanged.
__device__ __forceinline__ v16h unpack_nibbles_f16(unsigned wa, unsigned wb) {
  union { v16h v; v2h p[8]; } b;
#pragma unroll
  for (int m = 0; m < 4; ++m) {
    b.p[m]     = deq2(wa, m);   // halfs 2m,2m+1   <- nibbles m, m+4 of wa
    b.p[4 + m] = deq2(wb, m);   // halfs 8+2m,9+2m <- nibbles m, m+4 of wb
  }
  return b.v;
}

// Kernel 1: per-split partial dot products via f16 WMMA.
// grid.x = (OUT / 256) * nsplit ; block = 256 threads (8 waves).
// part[s * OUT + o] = sum_{k in slice s} x_f16[k] * w_int[k, o]
__global__ __launch_bounds__(BLOCK_THREADS)
void gptq4_wmma_partial(const float* __restrict__ x,
                        const int*   __restrict__ qw,
                        float*       __restrict__ part,
                        int IN, int OUT, int KS, int nsplit) {
  // LDS: [0, KS) = x slice as f16 (sigma-permuted per 8-group);
  //      [KS, KS+32) = zero block for A rows 1..15
  extern __shared__ _Float16 xs[];

  const int tid   = threadIdx.x;
  const int lane  = tid & 31;
  const int wave  = tid >> 5;
  const int hi    = lane >> 4;    // which half of the wave (K sub-range)
  const int nlo   = lane & 15;    // column-within-tile / M row index

  const int split = blockIdx.x % nsplit;
  const int tg    = blockIdx.x / nsplit;
  const int kbase = split * KS;

  // ---- Stage x slice into LDS as f16, permuted (0,4,1,5,2,6,3,7) / 8-group
  for (int i = tid * 8; i < KS; i += BLOCK_THREADS * 8) {
    const float4 v0 = *(const float4*)(x + kbase + i);
    const float4 v1 = *(const float4*)(x + kbase + i + 4);
    v8h h;
    h[0] = (_Float16)v0.x;  h[1] = (_Float16)v1.x;   // k+0, k+4
    h[2] = (_Float16)v0.y;  h[3] = (_Float16)v1.y;   // k+1, k+5
    h[4] = (_Float16)v0.z;  h[5] = (_Float16)v1.z;   // k+2, k+6
    h[6] = (_Float16)v0.w;  h[7] = (_Float16)v1.w;   // k+3, k+7
    *(v8h*)(xs + i) = h;
  }
  if (tid < 8) {                      // 64-byte zero block after the slice
    const v4h z = {};
    *(v4h*)(xs + KS + tid * 4) = z;
  }
  __syncthreads();

  // ---- Column assignment: 2 tiles (32 columns) per wave
  const int o0   = (tg * WAVES_PER_BLOCK + wave) * (TILES_PER_WAVE * 16);
  const int col0 = o0 + nlo;
  const int col1 = col0 + 16;

  v8f acc0 = {};
  v8f acc1 = {};

  const int r_base = kbase >> 3;   // qweight word-row of slice start
  const int steps  = KS >> 5;      // K advances 32 per WMMA

  // A-fragment LDS walk: row-0 lanes stream the x slice (A(K) = xs[32t+K]
  // with K = 16*hi + s, the contiguous k order B also uses); all other lanes park on the zero block with
  // stride 0 — no per-iteration cndmask zeroing needed.
  int       aoff   = (nlo == 0) ? (hi << 4) : KS;   // in halfs
  const int a_step = (nlo == 0) ? 32 : 0;

  // Strength-reduced row pointers: p0 walks rows rA = r_base + 4t + 2*hi,
  // p1 walks rows rA+1; col1 is a constant +64B (offset:64) off each.
  const long long stepw = 4LL * OUT;                 // words per K-step
  const int* p0 = qw + (long long)(r_base + 2 * hi) * OUT + col0;
  const int* p1 = p0 + OUT;

#pragma unroll 2
  for (int t = 0; t < steps; ++t) {
    const unsigned wa0 = (unsigned)p0[0];
    const unsigned wb0 = (unsigned)p1[0];
    const unsigned wa1 = (unsigned)p0[16];
    const unsigned wb1 = (unsigned)p1[16];

    // Prefetch two iterations ahead at WGP scope (pull into all cache
    // levels). col1 shares col0's 128B line; the two lane halves (hi)
    // jointly cover all four future rows. WGP-scope prefetch is
    // non-speculative, so guard against running past the K slice.
    if (t + 2 < steps) {
      const int* pf0 = p0 + 2 * stepw;
      const int* pf1 = p1 + 2 * stepw;
      asm volatile("global_prefetch_b8 %0, off" :: "v"(pf0));
      asm volatile("global_prefetch_b8 %0, off" :: "v"(pf1));
    }

    const v8h alo = *(const v8h*)(xs + aoff);        // ds_load_b128
    const v8h ahi = *(const v8h*)(xs + aoff + 8);    // ds_load_b128
    aoff += a_step;
    union { v16h v; v8h h[2]; } au;
    au.h[0] = alo; au.h[1] = ahi;
    const v16h a = au.v;

    const v16h b0 = unpack_nibbles_f16(wa0, wb0);
    const v16h b1 = unpack_nibbles_f16(wa1, wb1);

    acc0 = __builtin_amdgcn_wmma_f32_16x16x32_f16(
        false, a, false, b0, (short)0, acc0, false, false);
    acc1 = __builtin_amdgcn_wmma_f32_16x16x32_f16(
        false, a, false, b1, (short)0, acc1, false, false);

    p0 += stepw;
    p1 += stepw;
  }

  // D layout: lanes 0-15 hold N=lane, VGPR0 = row M=0 -> our matvec result.
  if (lane < 16) {
    part[(long long)split * OUT + col0] = acc0[0];
    part[(long long)split * OUT + col1] = acc1[0];
  }
}

// Kernel 2: y[o] = bias[o] + scales[o] * sum_s part[s][o] - zeros[o] * sum(x)
__global__ __launch_bounds__(BLOCK_THREADS)
void gptq4_finalize(const float* __restrict__ x,
                    const float* __restrict__ part,
                    const float* __restrict__ scales,
                    const float* __restrict__ zeros,
                    const float* __restrict__ bias,
                    float*       __restrict__ y,
                    int IN, int OUT, int nsplit) {
  __shared__ float red[BLOCK_THREADS];
  const int tid = threadIdx.x;

  float s = 0.0f;
  for (int i = tid; i < IN; i += BLOCK_THREADS) s += x[i];
  red[tid] = s;
  __syncthreads();
#pragma unroll
  for (int off = BLOCK_THREADS / 2; off > 0; off >>= 1) {
    if (tid < off) red[tid] += red[tid + off];
    __syncthreads();
  }
  const float xsum = red[0];

  const int o = blockIdx.x * BLOCK_THREADS + tid;
  if (o < OUT) {
    float d = 0.0f;
    for (int si = 0; si < nsplit; ++si) d += part[(long long)si * OUT + o];
    y[o] = bias[o] + scales[o] * d - zeros[o] * xsum;
  }
}

extern "C" void kernel_launch(void* const* d_in, const int* in_sizes, int n_in,
                              void* d_out, int out_size, void* d_ws, size_t ws_size,
                              hipStream_t stream) {
  const float* x      = (const float*)d_in[0];
  const int*   qw     = (const int*)d_in[1];
  const float* scales = (const float*)d_in[2];
  const float* zeros  = (const float*)d_in[3];
  const float* bias   = (const float*)d_in[4];
  float*       y      = (float*)d_out;
  float*       part   = (float*)d_ws;

  const int IN  = in_sizes[0];   // 8192
  const int OUT = in_sizes[4];   // 8192 (bias length)

  // Pick split-K: as deep as the workspace / divisibility allow (power of 2).
  int nsplit = MAX_SPLIT_K;
  while (nsplit > 1 &&
         (((size_t)nsplit * (size_t)OUT * sizeof(float)) > ws_size ||
          (IN % (nsplit * 64)) != 0)) {
    nsplit >>= 1;
  }
  const int KS = IN / nsplit;

  const int cols_per_block = WAVES_PER_BLOCK * TILES_PER_WAVE * 16;  // 256
  const int tgroups = OUT / cols_per_block;                          // 32

  dim3 blk(BLOCK_THREADS);
  dim3 g1(tgroups * nsplit);
  gptq4_wmma_partial<<<g1, blk, (size_t)KS * 2 + 64, stream>>>(
      x, qw, part, IN, OUT, KS, nsplit);

  dim3 g2((OUT + BLOCK_THREADS - 1) / BLOCK_THREADS);
  gptq4_finalize<<<g2, blk, 0, stream>>>(
      x, part, scales, zeros, bias, y, IN, OUT, nsplit);
}
